// Decoder_70583492542708
// MI455X (gfx1250) — compile-verified, matrix operand order checked by reading, not run on hardware
//
#include <hip/hip_runtime.h>
#include <math.h>

constexpr int NB = 32;      // batch
constexpr int NT = 32;      // timesteps
constexpr int NS = 400;     // source len
constexpr int NE = 512;     // embed
constexpr int NH = 512;     // hidden
constexpr int NV = 50000;   // vocab
constexpr int NX = 50;      // extra
constexpr int NVX = NV + NX;

typedef __attribute__((ext_vector_type(16))) _Float16 v16h;
typedef __attribute__((ext_vector_type(8)))  float    v8f;

union FragU { v16h v; uint4 q[2]; };

// 16x32 f16 WMMA fragment per CDNA5 16-bit A/B lane layout:
// lane l holds row (l&15); K halves {kb..kb+7} and {16+kb..16+kb+7},
// kb = (l>>4)*8.  Two 16-byte loads per lane.
__device__ inline v16h load_frag(const _Float16* __restrict__ base, int row,
                                 int ldk, int k0, int lane) {
  const int kb = (lane >> 4) << 3;
  const _Float16* p = base + (size_t)row * ldk + k0 + kb;
  FragU f;
  f.q[0] = *reinterpret_cast<const uint4*>(p);
  f.q[1] = *reinterpret_cast<const uint4*>(p + 16);
  return f.v;
}

__device__ inline void atomicMaxF(float* addr, float v) {
  if (v >= 0.f) atomicMax((int*)addr, __float_as_int(v));
  else          atomicMin((unsigned int*)addr, __float_as_uint(v));
}

__device__ inline float sigmoidf_(float x) { return 1.f / (1.f + __expf(-x)); }

// ---------------- setup kernels ----------------

__global__ void k_cvt(const float* __restrict__ src, int rows, int cols,
                      int sstride, int scoff,
                      _Float16* __restrict__ dst, int dstride, int dcoff) {
  long long n = (long long)rows * cols;
  for (long long i = blockIdx.x * (long long)blockDim.x + threadIdx.x; i < n;
       i += (long long)gridDim.x * blockDim.x) {
    int r = (int)(i / cols), c = (int)(i % cols);
    dst[(size_t)r * dstride + dcoff + c] = (_Float16)src[(size_t)r * sstride + scoff + c];
  }
}

__global__ void k_bsum(const float* __restrict__ a, const float* __restrict__ b,
                       float* __restrict__ o, int n) {
  int i = blockIdx.x * blockDim.x + threadIdx.x;
  if (i < n) o[i] = a[i] + b[i];
}

__global__ void k_init(const float* __restrict__ ctx0, const float* __restrict__ cov0,
                       float* __restrict__ c0, float* __restrict__ ctx,
                       float* __restrict__ cov, _Float16* __restrict__ A2) {
  int i = blockIdx.x * blockDim.x + threadIdx.x;
  if (i < NB * NH) {
    c0[i] = 0.f;
    ctx[i] = ctx0[i];
    A2[(i / NH) * (2 * NH) + NH + (i % NH)] = (_Float16)0.f;  // h_prev = 0
  }
  if (i < NB * NS) cov[i] = cov0[i];
}

// ---------------- generic WMMA GEMM, M = 32 ----------------
// C[m,n] = sum_k A[m,k] * W[n,k] (+ bias[n]); grid.x = N/128; 8 waves/block.
__global__ __launch_bounds__(256) void k_gemm32(const _Float16* __restrict__ A, int lda,
                                                const _Float16* __restrict__ W,
                                                const float* __restrict__ bias, int K,
                                                float* __restrict__ Cf32, int ldc,
                                                _Float16* __restrict__ Cf16,
                                                int ldc16, int c16off) {
  const int lane = threadIdx.x & 31;
  const int wave = threadIdx.x >> 5;
  const int n = blockIdx.x * 128 + wave * 16 + (lane & 15);
  v8f acc0 = {0.f,0.f,0.f,0.f,0.f,0.f,0.f,0.f};
  v8f acc1 = {0.f,0.f,0.f,0.f,0.f,0.f,0.f,0.f};
  const int a0row = lane & 15;
  const int a1row = 16 + (lane & 15);
  for (int k0 = 0; k0 < K; k0 += 32) {
    v16h w  = load_frag(W, n, K, k0, lane);
    v16h a0 = load_frag(A, a0row, lda, k0, lane);
    v16h a1 = load_frag(A, a1row, lda, k0, lane);
    acc0 = __builtin_amdgcn_wmma_f32_16x16x32_f16(false, a0, false, w, (short)0, acc0, false, false);
    acc1 = __builtin_amdgcn_wmma_f32_16x16x32_f16(false, a1, false, w, (short)0, acc1, false, false);
  }
  const float bb = bias ? bias[n] : 0.f;
  const int mh = (lane >> 4) << 3;
  #pragma unroll
  for (int r = 0; r < 8; ++r) {
    const float v0 = acc0[r] + bb;
    const float v1 = acc1[r] + bb;
    const int m0 = r + mh, m1 = 16 + r + mh;
    if (Cf32) {
      Cf32[(size_t)m0 * ldc + n] = v0;
      Cf32[(size_t)m1 * ldc + n] = v1;
    }
    if (Cf16) {
      Cf16[(size_t)m0 * ldc16 + c16off + n] = (_Float16)v0;
      Cf16[(size_t)m1 * ldc16 + c16off + n] = (_Float16)v1;
    }
  }
}

// enc_proj WMMA: M = B*S = 12800, N = 512, K = 512; one tile per wave.
__global__ __launch_bounds__(256) void k_encproj(const _Float16* __restrict__ A,
                                                 const _Float16* __restrict__ W,
                                                 float* __restrict__ C) {
  const int lane = threadIdx.x & 31;
  const int wave = threadIdx.x >> 5;
  const int gid  = blockIdx.x * 8 + wave;
  const int m0 = (gid >> 5) * 16;
  const int n0 = (gid & 31) * 16;
  v8f acc = {0.f,0.f,0.f,0.f,0.f,0.f,0.f,0.f};
  const int arow = m0 + (lane & 15);
  const int wrow = n0 + (lane & 15);
  for (int k0 = 0; k0 < NH; k0 += 32) {
    v16h a = load_frag(A, arow, NH, k0, lane);
    v16h w = load_frag(W, wrow, NH, k0, lane);
    acc = __builtin_amdgcn_wmma_f32_16x16x32_f16(false, a, false, w, (short)0, acc, false, false);
  }
  const int n  = n0 + (lane & 15);
  const int mh = (lane >> 4) << 3;
  #pragma unroll
  for (int r = 0; r < 8; ++r) C[(size_t)(m0 + r + mh) * NH + n] = acc[r];
}

// ---------------- per-step kernels ----------------

__global__ void k_pack(const float* __restrict__ emb, int t,
                       const float* __restrict__ ctx_prev, _Float16* __restrict__ A1) {
  int i = blockIdx.x * blockDim.x + threadIdx.x;
  if (i >= NB * 2 * NH) return;
  const int b = i / (2 * NH), c = i % (2 * NH);
  float v = (c < NE) ? emb[((size_t)b * NT + t) * NE + c] : ctx_prev[b * NH + (c - NE)];
  A1[i] = (_Float16)v;
}

__global__ void k_lstmelem(const float* __restrict__ gates, const float* __restrict__ c_prev,
                           float* __restrict__ hbuf, float* __restrict__ c_out,
                           _Float16* __restrict__ A4, _Float16* __restrict__ A2) {
  int i = blockIdx.x * blockDim.x + threadIdx.x;
  if (i >= NB * NH) return;
  const int b = i / NH, j = i % NH;
  const float gi = gates[(size_t)b * 4 * NH + j];
  const float gf = gates[(size_t)b * 4 * NH + NH + j];
  const float gg = gates[(size_t)b * 4 * NH + 2 * NH + j];
  const float go = gates[(size_t)b * 4 * NH + 3 * NH + j];
  const float cn = sigmoidf_(gf) * c_prev[i] + sigmoidf_(gi) * tanhf(gg);
  const float hn = sigmoidf_(go) * tanhf(cn);
  c_out[i] = cn;
  hbuf[i]  = hn;
  A4[(size_t)b * 2 * NH + j]      = (_Float16)hn;
  A2[(size_t)b * 2 * NH + NH + j] = (_Float16)hn;
}

// grid = B * 25 blocks; block: one b, 16 source positions; wave -> 2 positions.
__global__ __launch_bounds__(256) void k_energy(const float* __restrict__ dec,
                                                const float* __restrict__ encproj,
                                                const float* __restrict__ cov_prev,
                                                const float* __restrict__ wcov,
                                                const float* __restrict__ vattn,
                                                float* __restrict__ energy) {
  const int b  = blockIdx.x / 25;
  const int s0 = (blockIdx.x % 25) * 16;
  __shared__ float s_dec[NH];
  for (int i = threadIdx.x; i < NH; i += 256) s_dec[i] = dec[b * NH + i];
  __syncthreads();
  const int lane = threadIdx.x & 31, wave = threadIdx.x >> 5;
  #pragma unroll
  for (int i = 0; i < 2; ++i) {
    const int s = s0 + wave * 2 + i;
    const float cv = cov_prev[b * NS + s];
    const float* ep = encproj + ((size_t)b * NS + s) * NH;
    float acc = 0.f;
    for (int e = lane; e < NH; e += 32)
      acc += tanhf(s_dec[e] + ep[e] + cv * wcov[e]) * vattn[e];
    #pragma unroll
    for (int m = 1; m < 32; m <<= 1) acc += __shfl_xor(acc, m, 32);
    if (lane == 0) energy[b * NS + s] = acc;
  }
}

__global__ __launch_bounds__(256) void k_attnsm(const float* __restrict__ energy,
                                                const float* __restrict__ enc,
                                                const float* __restrict__ mask,
                                                const float* __restrict__ cov_prev,
                                                float* __restrict__ probs,
                                                float* __restrict__ cov_out,
                                                float* __restrict__ ctx_out,
                                                _Float16* __restrict__ A4) {
  const int b = blockIdx.x;
  __shared__ float s_en[NS];
  __shared__ float s_red[256];
  const int tid = threadIdx.x;
  for (int s = tid; s < NS; s += 256) s_en[s] = energy[b * NS + s];
  __syncthreads();
  float lm = -INFINITY;
  for (int s = tid; s < NS; s += 256) lm = fmaxf(lm, s_en[s]);
  s_red[tid] = lm; __syncthreads();
  for (int w = 128; w > 0; w >>= 1) { if (tid < w) s_red[tid] = fmaxf(s_red[tid], s_red[tid + w]); __syncthreads(); }
  const float gmax = s_red[0]; __syncthreads();
  float ls = 0.f;
  for (int s = tid; s < NS; s += 256) { float e = __expf(s_en[s] - gmax); s_en[s] = e; ls += e; }
  s_red[tid] = ls; __syncthreads();
  for (int w = 128; w > 0; w >>= 1) { if (tid < w) s_red[tid] += s_red[tid + w]; __syncthreads(); }
  const float den0 = s_red[0]; __syncthreads();
  float ls2 = 0.f;
  for (int s = tid; s < NS; s += 256) { float q = s_en[s] / den0 * mask[b * NS + s]; s_en[s] = q; ls2 += q; }
  s_red[tid] = ls2; __syncthreads();
  for (int w = 128; w > 0; w >>= 1) { if (tid < w) s_red[tid] += s_red[tid + w]; __syncthreads(); }
  const float den1 = s_red[0] + 1e-12f; __syncthreads();
  for (int s = tid; s < NS; s += 256) {
    float p = s_en[s] / den1;
    s_en[s] = p;
    probs[b * NS + s]   = p;
    cov_out[b * NS + s] = cov_prev[b * NS + s] + p;
  }
  __syncthreads();
  for (int e = tid; e < NH; e += 256) {
    float acc = 0.f;
    const float* ebase = enc + (size_t)b * NS * NH + e;
    for (int s = 0; s < NS; ++s) acc += s_en[s] * ebase[(size_t)s * NH];
    ctx_out[b * NH + e] = acc;
    A4[(size_t)b * 2 * NH + NH + e] = (_Float16)acc;
  }
}

__global__ __launch_bounds__(256) void k_pgen(const float* __restrict__ ctx,
                                              const float* __restrict__ h,
                                              const float* __restrict__ x,
                                              const float* __restrict__ Wpg,
                                              const float* __restrict__ bpg,
                                              float* __restrict__ pgen,
                                              float* __restrict__ row_max,
                                              float* __restrict__ row_sum) {
  const int b = blockIdx.x;
  __shared__ float s_red[256];
  const int tid = threadIdx.x;
  float acc = 0.f;
  for (int k = tid; k < 2 * NH + NE; k += 256) {
    float v = (k < NH) ? ctx[b * NH + k]
            : (k < 2 * NH) ? h[b * NH + (k - NH)]
            : x[b * NE + (k - 2 * NH)];
    acc += v * Wpg[k];
  }
  s_red[tid] = acc; __syncthreads();
  for (int w = 128; w > 0; w >>= 1) { if (tid < w) s_red[tid] += s_red[tid + w]; __syncthreads(); }
  if (tid == 0) {
    pgen[b]    = sigmoidf_(s_red[0] + bpg[0]);
    row_max[b] = -INFINITY;
    row_sum[b] = 0.f;
  }
}

__global__ __launch_bounds__(256) void k_vocab(const _Float16* __restrict__ A,
                                               const _Float16* __restrict__ Wv,
                                               const float* __restrict__ bv,
                                               float* __restrict__ logits,
                                               float* __restrict__ row_max) {
  const int lane = threadIdx.x & 31;
  const int wave = threadIdx.x >> 5;
  const int n  = blockIdx.x * 128 + wave * 16 + (lane & 15);
  const int nc = n < NV ? n : NV - 1;
  v8f acc0 = {0.f,0.f,0.f,0.f,0.f,0.f,0.f,0.f};
  v8f acc1 = {0.f,0.f,0.f,0.f,0.f,0.f,0.f,0.f};
  const int a0row = lane & 15;
  const int a1row = 16 + (lane & 15);
  for (int k0 = 0; k0 < NH; k0 += 32) {
    v16h w  = load_frag(Wv, nc, NH, k0, lane);
    v16h a0 = load_frag(A, a0row, NH, k0, lane);
    v16h a1 = load_frag(A, a1row, NH, k0, lane);
    acc0 = __builtin_amdgcn_wmma_f32_16x16x32_f16(false, a0, false, w, (short)0, acc0, false, false);
    acc1 = __builtin_amdgcn_wmma_f32_16x16x32_f16(false, a1, false, w, (short)0, acc1, false, false);
  }
  const float bias = bv[nc];
  const int mh = (lane >> 4) << 3;
  #pragma unroll
  for (int r = 0; r < 8; ++r) {
    float v0 = acc0[r] + bias;
    float v1 = acc1[r] + bias;
    const int m0r = r + mh;
    const int m1r = 16 + r + mh;
    if (n < NV) {
      logits[(size_t)m0r * NV + n] = v0;
      logits[(size_t)m1r * NV + n] = v1;
    }
    float r0 = v0, r1 = v1;
    #pragma unroll
    for (int msk = 1; msk < 16; msk <<= 1) {
      r0 = fmaxf(r0, __shfl_xor(r0, msk, 32));
      r1 = fmaxf(r1, __shfl_xor(r1, msk, 32));
    }
    if ((lane & 15) == 0) { atomicMaxF(&row_max[m0r], r0); atomicMaxF(&row_max[m1r], r1); }
  }
}

__global__ __launch_bounds__(256) void k_softexp(float* __restrict__ logits,
                                                 const float* __restrict__ row_max,
                                                 float* __restrict__ row_sum) {
  const int NCH = (NV + 2047) / 2048;
  const int row = blockIdx.x / NCH;
  const int base = (blockIdx.x % NCH) * 2048;
  const float m = row_max[row];
  __shared__ float s_red[256];
  float acc = 0.f;
  for (int i = threadIdx.x; i < 2048; i += 256) {
    const int v = base + i;
    if (v < NV) {
      float e = __expf(logits[(size_t)row * NV + v] - m);
      logits[(size_t)row * NV + v] = e;
      acc += e;
    }
  }
  s_red[threadIdx.x] = acc; __syncthreads();
  for (int w = 128; w > 0; w >>= 1) { if (threadIdx.x < w) s_red[threadIdx.x] += s_red[threadIdx.x + w]; __syncthreads(); }
  if (threadIdx.x == 0) atomicAdd(&row_sum[row], s_red[0]);
}

__global__ void k_final(const float* __restrict__ logits, const float* __restrict__ pgen,
                        const float* __restrict__ row_sum, int t, float* __restrict__ out) {
  const long long n = (long long)NB * NVX;
  for (long long i = blockIdx.x * (long long)blockDim.x + threadIdx.x; i < n;
       i += (long long)gridDim.x * blockDim.x) {
    const int b = (int)(i / NVX);
    const int col = (int)(i % NVX);
    float val = 0.f;
    if (col < NV) val = pgen[b] * logits[(size_t)b * NV + col] / row_sum[b];
    out[((size_t)b * NT + t) * NVX + col] = val;
  }
}

__global__ void k_scatter(const float* __restrict__ probs, const float* __restrict__ pgen,
                          const int* __restrict__ idx, int t, float* __restrict__ out) {
  const int i = blockIdx.x * blockDim.x + threadIdx.x;
  if (i >= NB * NS) return;
  const int b = i / NS;
  const int col = idx[i];
  atomicAdd(&out[((size_t)b * NT + t) * NVX + col], (1.f - pgen[b]) * probs[i]);
}

extern "C" void kernel_launch(void* const* d_in, const int* in_sizes, int n_in,
                              void* d_out, int out_size, void* d_ws, size_t ws_size,
                              hipStream_t stream) {
  (void)in_sizes; (void)n_in; (void)out_size; (void)ws_size;
  const float* emb   = (const float*)d_in[0];
  const float* enc   = (const float*)d_in[1];
  const float* mask  = (const float*)d_in[2];
  const float* ctx0  = (const float*)d_in[3];
  const float* cov0  = (const float*)d_in[4];
  const int*   esidx = (const int*)d_in[6];
  const float* Wih   = (const float*)d_in[7];
  const float* Whh   = (const float*)d_in[8];
  const float* bih   = (const float*)d_in[9];
  const float* bhh   = (const float*)d_in[10];
  const float* Wv    = (const float*)d_in[11];
  const float* bv    = (const float*)d_in[12];
  const float* Wxc   = (const float*)d_in[13];
  const float* bxc   = (const float*)d_in[14];
  const float* We    = (const float*)d_in[15];
  const float* be    = (const float*)d_in[16];
  const float* Wcov  = (const float*)d_in[17];
  const float* vattn = (const float*)d_in[18];
  const float* Wad   = (const float*)d_in[19];
  const float* bad   = (const float*)d_in[20];
  const float* Wpg   = (const float*)d_in[21];
  const float* bpg   = (const float*)d_in[22];
  float* out = (float*)d_out;

  char* ws = (char*)d_ws;
  size_t off = 0;
  auto alloc = [&](size_t bytes) -> char* {
    char* p = ws + off;
    off += (bytes + 255) & ~(size_t)255;
    return p;
  };
  _Float16* Wvh    = (_Float16*)alloc((size_t)NV * NH * 2);
  _Float16* ench   = (_Float16*)alloc((size_t)NB * NS * NH * 2);
  _Float16* Weenh  = (_Float16*)alloc((size_t)NH * NH * 2);
  _Float16* Wedech = (_Float16*)alloc((size_t)NH * NH * 2);
  _Float16* Wxch   = (_Float16*)alloc((size_t)NE * (NE + NH) * 2);
  _Float16* Wgh    = (_Float16*)alloc((size_t)4 * NH * (NE + NH) * 2);
  _Float16* Wadh   = (_Float16*)alloc((size_t)NH * 2 * NH * 2);
  float*    bg     = (float*)alloc((size_t)4 * NH * 4);
  float* encproj = (float*)alloc((size_t)NB * NS * NH * 4);
  float* logits  = (float*)alloc((size_t)NB * NV * 4);
  _Float16* A1    = (_Float16*)alloc((size_t)NB * 2 * NH * 2);
  _Float16* A2    = (_Float16*)alloc((size_t)NB * 2 * NH * 2);
  _Float16* A4    = (_Float16*)alloc((size_t)NB * 2 * NH * 2);
  _Float16* hid2h = (_Float16*)alloc((size_t)NB * NH * 2);
  float* gates  = (float*)alloc((size_t)NB * 4 * NH * 4);
  float* decbuf = (float*)alloc((size_t)NB * NH * 4);
  float* energy = (float*)alloc((size_t)NB * NS * 4);
  float* xbuf   = (float*)alloc((size_t)NB * NE * 4);
  float* hbuf   = (float*)alloc((size_t)NB * NH * 4);
  float* cbuf[2] = {(float*)alloc((size_t)NB * NH * 4), (float*)alloc((size_t)NB * NH * 4)};
  float* ctxb[2] = {(float*)alloc((size_t)NB * NH * 4), (float*)alloc((size_t)NB * NH * 4)};
  float* covb[2] = {(float*)alloc((size_t)NB * NS * 4), (float*)alloc((size_t)NB * NS * 4)};
  float* probs   = (float*)alloc((size_t)NB * NS * 4);
  float* pgen    = (float*)alloc(NB * 4);
  float* rowmax  = (float*)alloc(NB * 4);
  float* rowsum  = (float*)alloc(NB * 4);

  k_cvt<<<1024, 256, 0, stream>>>(Wv,  NV,      NH, NH,      0,  Wvh,    NH,      0);
  k_cvt<<<512,  256, 0, stream>>>(enc, NB * NS, NH, NH,      0,  ench,   NH,      0);
  k_cvt<<<128,  256, 0, stream>>>(We,  NH,      NH, 2 * NH,  NH, Weenh,  NH,      0);
  k_cvt<<<128,  256, 0, stream>>>(We,  NH,      NH, 2 * NH,  0,  Wedech, NH,      0);
  k_cvt<<<256,  256, 0, stream>>>(Wxc, NE, NE + NH, NE + NH, 0,  Wxch,   NE + NH, 0);
  k_cvt<<<512,  256, 0, stream>>>(Wih, 4 * NH, NE, NE,       0,  Wgh,    NE + NH, 0);
  k_cvt<<<512,  256, 0, stream>>>(Whh, 4 * NH, NH, NH,       0,  Wgh,    NE + NH, NE);
  k_cvt<<<256,  256, 0, stream>>>(Wad, NH, 2 * NH, 2 * NH,   0,  Wadh,   2 * NH,  0);
  k_bsum<<<(4 * NH + 255) / 256, 256, 0, stream>>>(bih, bhh, bg, 4 * NH);
  k_encproj<<<(NB * NS / 16) * (NH / 16) / 8, 256, 0, stream>>>(ench, Weenh, encproj);
  k_init<<<(NB * NH + 255) / 256, 256, 0, stream>>>(ctx0, cov0, cbuf[0], ctxb[0], covb[0], A2);

  for (int t = 0; t < NT; ++t) {
    const int pi = t & 1, ci = pi ^ 1;
    k_pack<<<(NB * 2 * NH + 255) / 256, 256, 0, stream>>>(emb, t, ctxb[pi], A1);
    k_gemm32<<<NE / 128, 256, 0, stream>>>(A1, 2 * NH, Wxch, bxc, NE + NH,
                                           xbuf, NE, A2, 2 * NH, 0);
    k_gemm32<<<(4 * NH) / 128, 256, 0, stream>>>(A2, 2 * NH, Wgh, bg, NE + NH,
                                                 gates, 4 * NH, (_Float16*)nullptr, 0, 0);
    k_lstmelem<<<(NB * NH + 255) / 256, 256, 0, stream>>>(gates, cbuf[pi], hbuf, cbuf[ci], A4, A2);
    k_gemm32<<<NH / 128, 256, 0, stream>>>(A4, 2 * NH, Wedech, be, NH,
                                           decbuf, NH, (_Float16*)nullptr, 0, 0);
    k_energy<<<NB * 25, 256, 0, stream>>>(decbuf, encproj, covb[pi], Wcov, vattn, energy);
    k_attnsm<<<NB, 256, 0, stream>>>(energy, enc, mask, covb[pi], probs, covb[ci], ctxb[ci], A4);
    k_gemm32<<<NH / 128, 256, 0, stream>>>(A4, 2 * NH, Wadh, bad, 2 * NH,
                                           (float*)nullptr, 0, hid2h, NH, 0);
    k_pgen<<<NB, 256, 0, stream>>>(ctxb[ci], hbuf, xbuf, Wpg, bpg, pgen, rowmax, rowsum);
    k_vocab<<<(NV + 127) / 128, 256, 0, stream>>>(hid2h, Wvh, bv, logits, rowmax);
    k_softexp<<<NB * ((NV + 2047) / 2048), 256, 0, stream>>>(logits, rowmax, rowsum);
    k_final<<<1024, 256, 0, stream>>>(logits, pgen, rowsum, t, out);
    k_scatter<<<(NB * NS + 255) / 256, 256, 0, stream>>>(probs, pgen, esidx, t, out);
  }
}
